// HeteroGNNs_45526653337611
// MI455X (gfx1250) — matrix operand order repaired and checked by reading, not compiled, not run on hardware
//
#include <hip/hip_runtime.h>

#define HID 768
#define NNODE 50000
#define NEG_SLOPE 0.2f

typedef _Float16 v16h __attribute__((ext_vector_type(16)));
typedef _Float16 v8h  __attribute__((ext_vector_type(8)));
typedef float    v8f  __attribute__((ext_vector_type(8)));

__device__ __forceinline__ void wait_asynccnt_0() {
#if __has_builtin(__builtin_amdgcn_s_wait_asynccnt)
  __builtin_amdgcn_s_wait_asynccnt(0);
#else
  asm volatile("s_wait_asynccnt 0" ::: "memory");
#endif
}
__device__ __forceinline__ void wait_asynccnt_1() {
#if __has_builtin(__builtin_amdgcn_s_wait_asynccnt)
  __builtin_amdgcn_s_wait_asynccnt(1);
#else
  asm volatile("s_wait_asynccnt 1" ::: "memory");
#endif
}

// Async DMA: 16 bytes global -> LDS, tracked by ASYNCcnt (no VGPR round-trip).
__device__ __forceinline__ void async_copy_b128(unsigned lds_addr,
                                                const void* gaddr) {
  asm volatile("global_load_async_to_lds_b128 %0, %1, off"
               :: "v"(lds_addr), "v"((unsigned long long)gaddr)
               : "memory");
}

// ---------------------------------------------------------------------------
// Pre-pass: f32 -> f16 bulk convert (8 elems/thread, 128-bit ld/st)
// ---------------------------------------------------------------------------
__global__ __launch_bounds__(256)
void f32_to_f16_kernel(const float* __restrict__ src,
                       _Float16* __restrict__ dst, size_t n) {
  size_t i = (blockIdx.x * (size_t)blockDim.x + threadIdx.x) * 8;
  if (i >= n) return;
  float4 a = *(const float4*)(src + i);
  float4 b = *(const float4*)(src + i + 4);
  v8h o = { (_Float16)a.x, (_Float16)a.y, (_Float16)a.z, (_Float16)a.w,
            (_Float16)b.x, (_Float16)b.y, (_Float16)b.z, (_Float16)b.w };
  *(v8h*)(dst + i) = o;
}

// ---------------------------------------------------------------------------
// Pre-pass: retile W[K x 768] (f32) into WMMA-B fragment layout (f16):
//   Bt[((k/32)*768 + n)*32 + (k%32)]
// so a B fragment (fixed n, K=kbB..kbB+15 of k-tile t) is 32 contiguous bytes.
// ---------------------------------------------------------------------------
__global__ __launch_bounds__(256)
void retile_w_kernel(const float* __restrict__ W,
                     _Float16* __restrict__ Bt, int K) {
  int idx = blockIdx.x * blockDim.x + threadIdx.x;
  if (idx >= K * HID) return;
  int kr = idx / HID, n = idx - kr * HID;
  Bt[(((size_t)(kr >> 5)) * HID + n) * 32 + (kr & 31)] = (_Float16)W[idx];
}

// ---------------------------------------------------------------------------
// Tiled WMMA GEMM: H[N x 768] = Xh[N x K] * W (as Bt), f16 MACs, f32 acc.
// Block = 256 threads = 8 waves; block tile 64(M) x 128(N); wave tile 16x64.
// A: double-buffered LDS, filled by async global->LDS DMA (ASYNCcnt),
//    prefetching tile t+1 while tile t feeds the WMMAs.  K is a template
//    parameter and the last tile is peeled, so the pipeline is branch-free
//    and fully unrolled (buffer selects fold to constants).
// B: fragment-ready in global (L2-resident), read as 32B v16h loads.
// ---------------------------------------------------------------------------
template <int K>
__global__ __launch_bounds__(256)
void gemm_hs_kernel(const _Float16* __restrict__ Xh,
                    const _Float16* __restrict__ Bt,
                    float* __restrict__ H, int N) {
  constexpr int nTiles = K / 32;
  __shared__ _Float16 lds_a[2][64 * 32];   // double-buffered A tile

  const int tid  = threadIdx.x;
  const int lane = tid & 31;
  const int wave = tid >> 5;
  const int wm   = wave & 3;             // M sub-tile (16 rows each)
  const int wn   = wave >> 2;            // N half (64 cols each)
  const int blockM = blockIdx.x * 64;
  const int blockN = blockIdx.y * 128;

  // A staging coords: thread DMAs 8 halves (16B) of row ar, cols ac..ac+7
  const int ar = tid >> 2;               // 0..63
  const int ac = (tid & 3) * 8;          // 0,8,16,24
  int gr = blockM + ar; if (gr >= N) gr = N - 1;   // clamp (no branches)
  const _Float16* asrc = Xh + (size_t)gr * K + ac;

  // LDS byte addresses for this thread's DMA targets (flat addr low 32 bits)
  const unsigned loff0 = (unsigned)(unsigned long long)&lds_a[0][ar * 32 + ac];
  const unsigned loff1 = (unsigned)(unsigned long long)&lds_a[1][ar * 32 + ac];

  // Fragment addressing (contiguous k = 16*(lane/16) + s for A and B, wave32)
  const int kb   = (lane >= 16) ? 16 : 0;    // A halves: K=kb..kb+15
  const int kbB  = (lane >= 16) ? 16 : 0;    // B halves: K=kbB..kbB+15
  const int mrow = wm * 16 + (lane & 15);
  const int ncol = lane & 15;

  v8f acc[4] = {};

  auto mma_step = [&](int t) {
    const _Float16* abuf = lds_a[t & 1];
    v8h alo = *(const v8h*)&abuf[mrow * 32 + kb];
    v8h ahi = *(const v8h*)&abuf[mrow * 32 + kb + 8];
    v16h afrag = __builtin_shufflevector(alo, ahi,
        0, 1, 2, 3, 4, 5, 6, 7, 8, 9, 10, 11, 12, 13, 14, 15);
    #pragma unroll
    for (int j = 0; j < 4; ++j) {
      int col = blockN + wn * 64 + j * 16 + ncol;
      v16h bfrag = *(const v16h*)(Bt + ((size_t)t * HID + col) * 32 + kbB);
      acc[j] = __builtin_amdgcn_wmma_f32_16x16x32_f16(
          false, afrag, false, bfrag, (short)0, acc[j], false, false);
    }
  };

  // Prefetch tile 0 into buffer 0
  async_copy_b128(loff0, asrc);

  #pragma unroll
  for (int t = 0; t < nTiles - 1; ++t) {
    // Prefetch next tile into the other buffer, then wait only for the
    // current tile (async ops complete in order -> the 1 outstanding is next).
    async_copy_b128(((t + 1) & 1) ? loff1 : loff0, asrc + (t + 1) * 32);
    wait_asynccnt_1();
    __syncthreads();
    mma_step(t);
    __syncthreads();
  }
  wait_asynccnt_0();
  __syncthreads();
  mma_step(nTiles - 1);

  // Epilogue: D layout — VGPR r: m = r + (lane>=16?8:0); n = lane&15
  const int mofs = (lane >= 16) ? 8 : 0;
  #pragma unroll
  for (int j = 0; j < 4; ++j) {
    int col = blockN + wn * 64 + j * 16 + ncol;
    #pragma unroll
    for (int r = 0; r < 8; ++r) {
      int row = blockM + wm * 16 + mofs + r;
      if (row < N) H[(size_t)row * HID + col] = acc[j][r];
    }
  }
}

// ---------------------------------------------------------------------------
// Row dot products: out[i] = dot(M[i, 0:K], v[0:K]); one wave per row.
// ---------------------------------------------------------------------------
__device__ inline float wave_reduce(float v) {
  #pragma unroll
  for (int off = 16; off > 0; off >>= 1) v += __shfl_down(v, off, 32);
  return v;
}

__global__ __launch_bounds__(256)
void dot_rows_kernel(const float* __restrict__ M, const float* __restrict__ vec,
                     float* __restrict__ out, int N, int K) {
  int row  = (int)((blockIdx.x * (size_t)blockDim.x + threadIdx.x) >> 5);
  int lane = threadIdx.x & 31;
  if (row >= N) return;
  const float* r = M + (size_t)row * K;
  float s = 0.f;
  for (int c = lane; c < K; c += 32) s += r[c] * vec[c];
  s = wave_reduce(s);
  if (lane == 0) out[row] = s;
}

// ---------------------------------------------------------------------------
// Segment softmax scaffolding
// ---------------------------------------------------------------------------
__global__ void init_seg_kernel(unsigned* __restrict__ maxb,
                                float* __restrict__ sumb, int n) {
  int i = blockIdx.x * blockDim.x + threadIdx.x;
  if (i < n) { maxb[i] = 0u; sumb[i] = 0.f; }
}

__device__ inline void edge_pair(const int* ei, int E, int e, int& src, int& dst) {
  if (e < E) { src = ei[e]; dst = ei[E + e]; }
  else       { src = dst = e - E; }        // appended self loop
}

__device__ inline float leaky(float x) { return x > 0.f ? x : NEG_SLOPE * x; }

__global__ void edge_max_kernel(const int* __restrict__ ei, int E, int nloop,
                                const float* __restrict__ as,
                                const float* __restrict__ ad,
                                unsigned* __restrict__ maxb) {
  int e = blockIdx.x * blockDim.x + threadIdx.x;
  if (e >= E + nloop) return;
  int src, dst; edge_pair(ei, E, e, src, dst);
  float l = leaky(as[src] + ad[dst]);
  unsigned u = __float_as_uint(l);
  unsigned enc = (u & 0x80000000u) ? ~u : (u | 0x80000000u);
  atomicMax(maxb + dst, enc);
}

__global__ void edge_exp_kernel(const int* __restrict__ ei, int E, int nloop,
                                const float* __restrict__ as,
                                const float* __restrict__ ad,
                                const unsigned* __restrict__ maxb,
                                float* __restrict__ sumb,
                                float* __restrict__ ebuf) {
  int e = blockIdx.x * blockDim.x + threadIdx.x;
  if (e >= E + nloop) return;
  int src, dst; edge_pair(ei, E, e, src, dst);
  float l = leaky(as[src] + ad[dst]);
  unsigned u = maxb[dst];
  float m = 0.f;
  if (u != 0u)
    m = (u & 0x80000000u) ? __uint_as_float(u & 0x7FFFFFFFu)
                          : __uint_as_float(~u);
  float ev = expf(l - m);
  ebuf[e] = ev;
  __hip_atomic_fetch_add(sumb + dst, ev, __ATOMIC_RELAXED,
                         __HIP_MEMORY_SCOPE_AGENT);
}

__global__ __launch_bounds__(256)
void edge_scatter_kernel(const int* __restrict__ ei, int E,
                         const float* __restrict__ hs,
                         const float* __restrict__ ebuf,
                         const float* __restrict__ sumb,
                         float* __restrict__ out) {
  int e = blockIdx.x;
  int src, dst; edge_pair(ei, E, e, src, dst);
  float w = ebuf[e] / (sumb[dst] + 1e-16f);
  const float* hrow = hs + (size_t)src * HID;
  float* orow = out + (size_t)dst * HID;
  for (int c = threadIdx.x; c < HID; c += blockDim.x)
    __hip_atomic_fetch_add(orow + c, hrow[c] * w, __ATOMIC_RELAXED,
                           __HIP_MEMORY_SCOPE_AGENT);
}

// ---------------------------------------------------------------------------
// Output init (bias sums) and final ReLU
// ---------------------------------------------------------------------------
__global__ void init_out_kernel(float* __restrict__ out,
                                const float* __restrict__ b_dd,
                                const float* __restrict__ b_rd,
                                const float* __restrict__ b_dt,
                                const float* __restrict__ b_tt) {
  size_t i = blockIdx.x * (size_t)blockDim.x + threadIdx.x;
  const size_t half = (size_t)NNODE * HID;
  if (i >= 2 * half) return;
  int c = (int)(i % HID);
  out[i] = (i < half) ? (b_dd[c] + b_rd[c]) : (b_dt[c] + b_tt[c]);
}

__global__ void relu_kernel(float* __restrict__ out, size_t total) {
  size_t i = blockIdx.x * (size_t)blockDim.x + threadIdx.x;
  if (i < total) out[i] = fmaxf(out[i], 0.f);
}

// ---------------------------------------------------------------------------
// Host orchestration
// ---------------------------------------------------------------------------
extern "C" void kernel_launch(void* const* d_in, const int* in_sizes, int n_in,
                              void* d_out, int out_size, void* d_ws, size_t ws_size,
                              hipStream_t stream) {
  const float* x_drug   = (const float*)d_in[0];   // [50000 x 256]
  const float* x_target = (const float*)d_in[1];   // [50000 x 512]
  const int* ei_dd = (const int*)d_in[2];
  const int* ei_dt = (const int*)d_in[3];
  const int* ei_rd = (const int*)d_in[4];
  const int* ei_tt = (const int*)d_in[5];
  const int E = in_sizes[2] / 2;

  float* out = (float*)d_out;
  const size_t half = (size_t)NNODE * HID;
  const size_t nXd = (size_t)NNODE * 256;
  const size_t nXt = (size_t)NNODE * 512;

  // Workspace layout (relations processed sequentially -> one hs buffer)
  char* ws = (char*)d_ws;
  size_t o = 0;
  float*     hs    = (float*)(ws + o);     o += half * sizeof(float);
  _Float16*  xh_d  = (_Float16*)(ws + o);  o += nXd * sizeof(_Float16);
  _Float16*  xh_t  = (_Float16*)(ws + o);  o += nXt * sizeof(_Float16);
  _Float16*  Bt    = (_Float16*)(ws + o);  o += (size_t)512 * HID * sizeof(_Float16);
  float*     a_s   = (float*)(ws + o);     o += (size_t)NNODE * sizeof(float);
  float*     a_d   = (float*)(ws + o);     o += (size_t)NNODE * sizeof(float);
  unsigned*  maxb  = (unsigned*)(ws + o);  o += (size_t)NNODE * sizeof(unsigned);
  float*     sumb  = (float*)(ws + o);     o += (size_t)NNODE * sizeof(float);
  float*     ebuf  = (float*)(ws + o);     o += (size_t)(E + NNODE) * sizeof(float);
  float*     wvec  = (float*)(ws + o);     o += 512 * sizeof(float);

  // Pre-convert node features to f16 (once)
  f32_to_f16_kernel<<<(unsigned)((nXd / 8 + 255) / 256), 256, 0, stream>>>(
      x_drug, xh_d, nXd);
  f32_to_f16_kernel<<<(unsigned)((nXt / 8 + 255) / 256), 256, 0, stream>>>(
      x_target, xh_t, nXt);

  // Init output with summed biases: drug <- b_dd+b_rd ; target <- b_dt+b_tt
  {
    size_t total = 2 * half;
    init_out_kernel<<<dim3((unsigned)((total + 255) / 256)), 256, 0, stream>>>(
        out, (const float*)d_in[10], (const float*)d_in[20],
             (const float*)d_in[15], (const float*)d_in[25]);
  }

  struct Rel {
    const _Float16* xs; int Ks;   // source features (f16)
    const float* xd; int Kd;      // dest features (f32, for a_d matvec)
    const int* ei; int self;      // edges, append self loops?
    size_t outOfs;                // 0 = drug part, half = target part
    int pbase;                    // d_in index of W_s
  };
  const Rel rels[4] = {
    { xh_d, 256, x_drug,   256, ei_dd, 1, 0,    6  },  // dd -> drug
    { xh_d, 256, x_target, 512, ei_dt, 0, half, 11 },  // dt -> target
    { xh_t, 512, x_drug,   256, ei_rd, 0, 0,    16 },  // rd -> drug
    { xh_t, 512, x_target, 512, ei_tt, 1, half, 21 },  // tt -> target
  };

  for (int r = 0; r < 4; ++r) {
    const Rel& R = rels[r];
    const float* W_s   = (const float*)d_in[R.pbase + 0];
    const float* W_d   = (const float*)d_in[R.pbase + 1];
    const float* att_s = (const float*)d_in[R.pbase + 2];
    const float* att_d = (const float*)d_in[R.pbase + 3];

    // 1) Retile W_s into fragment-ready f16 layout, then hs = x_src @ W_s
    retile_w_kernel<<<(R.Ks * HID + 255) / 256, 256, 0, stream>>>(W_s, Bt, R.Ks);
    dim3 ggrid((NNODE + 63) / 64, HID / 128);
    if (R.Ks == 256)
      gemm_hs_kernel<256><<<ggrid, 256, 0, stream>>>(R.xs, Bt, hs, NNODE);
    else
      gemm_hs_kernel<512><<<ggrid, 256, 0, stream>>>(R.xs, Bt, hs, NNODE);

    // 2) wvec = W_d @ att_d   (rank-1 fusion: avoids the full hd GEMM)
    dot_rows_kernel<<<(R.Kd + 7) / 8, 256, 0, stream>>>(W_d, att_d, wvec, R.Kd, HID);

    // 3) a_s[i] = dot(hs[i], att_s);  a_d[i] = dot(x_dst[i], wvec)
    dot_rows_kernel<<<(NNODE + 7) / 8, 256, 0, stream>>>(hs, att_s, a_s, NNODE, HID);
    dot_rows_kernel<<<(NNODE + 7) / 8, 256, 0, stream>>>(R.xd, wvec, a_d, NNODE, R.Kd);

    // 4) segment softmax over destinations
    init_seg_kernel<<<(NNODE + 255) / 256, 256, 0, stream>>>(maxb, sumb, NNODE);
    int tot = E + (R.self ? NNODE : 0);
    edge_max_kernel<<<(tot + 255) / 256, 256, 0, stream>>>(R.ei, E, tot - E,
                                                           a_s, a_d, maxb);
    edge_exp_kernel<<<(tot + 255) / 256, 256, 0, stream>>>(R.ei, E, tot - E,
                                                           a_s, a_d, maxb, sumb, ebuf);

    // 5) out[dst] += hs[src] * alpha   (coalesced row scatter, HW f32 fadd)
    edge_scatter_kernel<<<tot, 256, 0, stream>>>(R.ei, E, hs, ebuf, sumb,
                                                 out + R.outOfs);
  }

  // Final ReLU over both node sets
  {
    size_t total = 2 * half;
    relu_kernel<<<dim3((unsigned)((total + 255) / 256)), 256, 0, stream>>>(out, total);
  }
}
